// VAE_13108240187955
// MI455X (gfx1250) — matrix operand order repaired and checked by reading, not compiled, not run on hardware
//
#include <hip/hip_runtime.h>

#define HID    256
#define GATES  1024
#define MAXLEN 25
#define NVOCAB 28

typedef __attribute__((ext_vector_type(16))) _Float16 v16h;
typedef __attribute__((ext_vector_type(8)))  _Float16 v8h;
typedef __attribute__((ext_vector_type(8)))  float    v8f;

__device__ __forceinline__ v16h cat16(v8h lo, v8h hi) {
  return __builtin_shufflevector(lo, hi, 0,1,2,3,4,5,6,7,8,9,10,11,12,13,14,15);
}

__device__ __forceinline__ float sigm(float x) { return 1.0f / (1.0f + __expf(-x)); }

// g(1024) = Wih(1024x256,f16) * x(256,f16) + Whh(1024x256,f16) * h(256,f16)
// via chained V_WMMA_F32_16X16X32_F16. B operand = vector chunk broadcast to
// all 16 columns, so every column of D holds the same 16 partial sums; lanes
// with (lane%16)==0 write the 16 rows of each tile to LDS.
// All 8 waves of the block participate (8 row-tiles each); EXEC is all-ones
// through every WMMA (divergence only in the final write).
__device__ __forceinline__ void matvec1024_wmma(
    const _Float16* __restrict__ Wih, const _Float16* __restrict__ Whh,
    const _Float16* xs, const _Float16* hs, float* g)
{
  const int tid  = threadIdx.x;
  const int wave = tid >> 5;
  const int lane = tid & 31;
  const int m    = lane & 15;   // A-fragment row
  const int hlf  = lane >> 4;   // half-wave selects K sub-block

  for (int tile = wave; tile < 64; tile += 8) {
    const int row0 = tile << 4;
    v8f acc = {};
    const _Float16* arow = Wih + (size_t)(row0 + m) * 256;
    #pragma unroll
    for (int k0 = 0; k0 < 256; k0 += 32) {
      v8h  lo = *(const v8h*)(arow + k0 + hlf * 8);        // K = h*8 .. h*8+7
      v8h  hi = *(const v8h*)(arow + k0 + 16 + hlf * 8);   // K = 16+h*8 ..
      v16h a  = cat16(lo, hi);
      v16h b  = cat16(*(const v8h*)(xs + k0 + hlf * 8), *(const v8h*)(xs + k0 + 16 + hlf * 8));        // K matches A, all N
      acc = __builtin_amdgcn_wmma_f32_16x16x32_f16(false, a, false, b,
                                                   (short)0, acc, false, false);
    }
    arow = Whh + (size_t)(row0 + m) * 256;
    #pragma unroll
    for (int k0 = 0; k0 < 256; k0 += 32) {
      v8h  lo = *(const v8h*)(arow + k0 + hlf * 8);
      v8h  hi = *(const v8h*)(arow + k0 + 16 + hlf * 8);
      v16h a  = cat16(lo, hi);
      v16h b  = cat16(*(const v8h*)(hs + k0 + hlf * 8), *(const v8h*)(hs + k0 + 16 + hlf * 8));
      acc = __builtin_amdgcn_wmma_f32_16x16x32_f16(false, a, false, b,
                                                   (short)0, acc, false, false);
    }
    if (m == 0) {                       // lanes 0 and 16: M = r + 8*hlf
      #pragma unroll
      for (int r = 0; r < 8; ++r) g[row0 + r + hlf * 8] = acc[r];
    }
  }
}

// One-time f32 -> f16 conversion of the four recurrent weight matrices.
__global__ void __launch_bounds__(256) vae_cvt_weights(
    const float* __restrict__ a, const float* __restrict__ b,
    const float* __restrict__ c, const float* __restrict__ d,
    _Float16* __restrict__ out)
{
  const int i = blockIdx.x * blockDim.x + threadIdx.x;
  const int N = 1024 * 256;
  if (i < N) {
    out[i]         = (_Float16)a[i];
    out[N + i]     = (_Float16)b[i];
    out[2 * N + i] = (_Float16)c[i];
    out[3 * N + i] = (_Float16)d[i];
  }
}

__global__ void __launch_bounds__(256) vae_lstm_persistent(
    const int* __restrict__ data, const int* __restrict__ data_c,
    const int* __restrict__ target_c,
    const float* __restrict__ cond_emb, const float* __restrict__ enc_emb,
    const _Float16* __restrict__ w16,
    const float* __restrict__ enc_bih, const float* __restrict__ enc_bhh,
    const float* __restrict__ hmu_W, const float* __restrict__ hmu_b,
    const float* __restrict__ cmu_W, const float* __restrict__ cmu_b,
    const float* __restrict__ fc1_W, const float* __restrict__ fc1_b,
    const float* __restrict__ fc2_W, const float* __restrict__ fc2_b,
    const float* __restrict__ dec_emb,
    const float* __restrict__ dec_bih, const float* __restrict__ dec_bhh,
    const float* __restrict__ out_W, const float* __restrict__ out_b,
    float* __restrict__ out)
{
  __shared__ __align__(64) float    g[GATES];
  __shared__ __align__(64) float    bsum[GATES];
  __shared__ __align__(64) float    hbuf[HID];
  __shared__ __align__(64) float    cbuf[HID];
  __shared__ __align__(64) _Float16 x16[HID];
  __shared__ __align__(64) _Float16 h16[HID];
  __shared__ float mu_h[32];
  __shared__ float mu_c[32];
  __shared__ float vec40h[40];
  __shared__ float vec40c[40];
  __shared__ float lg[NVOCAB];
  __shared__ int   tok;

  const int j = threadIdx.x;
  const size_t N = (size_t)1024 * 256;
  const _Float16* encWih = w16;
  const _Float16* encWhh = w16 + N;
  const _Float16* decWih = w16 + 2 * N;
  const _Float16* decWhh = w16 + 3 * N;

  // ---- init: h0 = c0 = [zeros(248), cond_emb[data_c]] ----
  const int dcls = data_c[0];
  float h0 = (j < HID - 8) ? 0.0f : cond_emb[dcls * 8 + (j - (HID - 8))];
  hbuf[j] = h0; cbuf[j] = h0; h16[j] = (_Float16)h0;
  #pragma unroll
  for (int kk = 0; kk < 4; ++kk)
    bsum[kk * 256 + j] = enc_bih[kk * 256 + j] + enc_bhh[kk * 256 + j];
  __syncthreads();

  // ---- encoder: 16 sequential LSTM steps ----
  for (int t = 0; t < 16; ++t) {
    const int w = data[t];
    x16[j] = (_Float16)enc_emb[w * 256 + j];
    __syncthreads();
    matvec1024_wmma(encWih, encWhh, x16, h16, g);
    __syncthreads();
    float gi = g[j]       + bsum[j];
    float gf = g[256 + j] + bsum[256 + j];
    float gc = g[512 + j] + bsum[512 + j];
    float go = g[768 + j] + bsum[768 + j];
    float cn = sigm(gf) * cbuf[j] + sigm(gi) * tanhf(gc);
    float hn = sigm(go) * tanhf(cn);
    cbuf[j] = cn; hbuf[j] = hn; h16[j] = (_Float16)hn;
    __syncthreads();
  }

  // ---- latent projections: h_mu / c_mu (32x256 matvecs, VALU) ----
  if (j < 32) {
    float s = hmu_b[j];
    for (int k = 0; k < 256; ++k) s += hmu_W[j * 256 + k] * hbuf[k];
    mu_h[j] = s;
  } else if (j < 64) {
    const int o = j - 32;
    float s = cmu_b[o];
    for (int k = 0; k < 256; ++k) s += cmu_W[o * 256 + k] * cbuf[k];
    mu_c[o] = s;
  }
  __syncthreads();
  if (j < 32)      { vec40h[j] = mu_h[j]; vec40c[j] = mu_c[j]; }
  else if (j < 40) { const float tc = cond_emb[target_c[0] * 8 + (j - 32)];
                     vec40h[j] = tc; vec40c[j] = tc; }
  __syncthreads();
  {
    float sh = fc1_b[j], sc = fc2_b[j];
    for (int k = 0; k < 40; ++k) {
      sh += fc1_W[j * 40 + k] * vec40h[k];
      sc += fc2_W[j * 40 + k] * vec40c[k];
    }
    hbuf[j] = sh; cbuf[j] = sc; h16[j] = (_Float16)sh;
  }
  #pragma unroll
  for (int kk = 0; kk < 4; ++kk)
    bsum[kk * 256 + j] = dec_bih[kk * 256 + j] + dec_bhh[kk * 256 + j];
  if (j == 0) tok = 0;   // SOS
  __syncthreads();

  // ---- decoder: 25 steps with argmax token feedback ----
  for (int t = 0; t < MAXLEN; ++t) {
    x16[j] = (_Float16)fmaxf(dec_emb[tok * 256 + j], 0.0f);
    __syncthreads();
    matvec1024_wmma(decWih, decWhh, x16, h16, g);
    __syncthreads();
    float gi = g[j]       + bsum[j];
    float gf = g[256 + j] + bsum[256 + j];
    float gc = g[512 + j] + bsum[512 + j];
    float go = g[768 + j] + bsum[768 + j];
    float cn = sigm(gf) * cbuf[j] + sigm(gi) * tanhf(gc);
    float hn = sigm(go) * tanhf(cn);
    cbuf[j] = cn; hbuf[j] = hn; h16[j] = (_Float16)hn;
    __syncthreads();
    if (j < NVOCAB) {
      float s = out_b[j];
      for (int k = 0; k < 256; ++k) s += out_W[j * 256 + k] * hbuf[k];
      lg[j] = s;
      out[t * NVOCAB + j] = s;
    }
    __syncthreads();
    if (j == 0) {
      int best = 0; float bv = lg[0];
      for (int k = 1; k < NVOCAB; ++k) if (lg[k] > bv) { bv = lg[k]; best = k; }
      tok = best;
      out[MAXLEN * NVOCAB + t] = (float)best;   // toks appended after logits
    }
    __syncthreads();
  }
}

extern "C" void kernel_launch(void* const* d_in, const int* in_sizes, int n_in,
                              void* d_out, int out_size, void* d_ws, size_t ws_size,
                              hipStream_t stream)
{
  (void)in_sizes; (void)n_in; (void)out_size; (void)ws_size;
  const int*   data     = (const int*)  d_in[0];
  const int*   data_c   = (const int*)  d_in[1];
  const int*   target_c = (const int*)  d_in[2];
  const float* cond_emb = (const float*)d_in[3];
  const float* enc_emb  = (const float*)d_in[4];
  const float* enc_Wih  = (const float*)d_in[5];
  const float* enc_Whh  = (const float*)d_in[6];
  const float* enc_bih  = (const float*)d_in[7];
  const float* enc_bhh  = (const float*)d_in[8];
  const float* hmu_W    = (const float*)d_in[9];
  const float* hmu_b    = (const float*)d_in[10];
  const float* cmu_W    = (const float*)d_in[11];
  const float* cmu_b    = (const float*)d_in[12];
  const float* fc1_W    = (const float*)d_in[13];
  const float* fc1_b    = (const float*)d_in[14];
  const float* fc2_W    = (const float*)d_in[15];
  const float* fc2_b    = (const float*)d_in[16];
  const float* dec_emb  = (const float*)d_in[17];
  const float* dec_Wih  = (const float*)d_in[18];
  const float* dec_Whh  = (const float*)d_in[19];
  const float* dec_bih  = (const float*)d_in[20];
  const float* dec_bhh  = (const float*)d_in[21];
  const float* out_W    = (const float*)d_in[22];
  const float* out_b    = (const float*)d_in[23];
  float*       out      = (float*)d_out;

  _Float16* w16 = (_Float16*)d_ws;   // 4 x 1024 x 256 f16 = 2 MB
  const int N = 1024 * 256;
  vae_cvt_weights<<<(N + 255) / 256, 256, 0, stream>>>(
      enc_Wih, enc_Whh, dec_Wih, dec_Whh, w16);

  vae_lstm_persistent<<<1, 256, 0, stream>>>(
      data, data_c, target_c, cond_emb, enc_emb, w16,
      enc_bih, enc_bhh, hmu_W, hmu_b, cmu_W, cmu_b,
      fc1_W, fc1_b, fc2_W, fc2_b,
      dec_emb, dec_bih, dec_bhh, out_W, out_b, out);
}
